// GlobalInteractionModule_82918638617236
// MI455X (gfx1250) — matrix operand order repaired and checked by reading, not compiled, not run on hardware
//
#include <hip/hip_runtime.h>
#include <hip/hip_bf16.h>

// ---------------------------------------------------------------------------
// GlobalInteractionModule: cross-community multi-head attention + out-proj +
// residual + layernorm, built around v_wmma_f32_16x16x32_bf16 (CDNA5).
// ---------------------------------------------------------------------------

#define NTOK   8192
#define DMODEL 1024
#define NHEAD  8
#define DHEAD  128

typedef __bf16 bf16;
typedef __attribute__((ext_vector_type(16))) __bf16 v16bf;
typedef __attribute__((ext_vector_type(8)))  float  v8f;
typedef __attribute__((ext_vector_type(4)))  unsigned int u32x4;
typedef __attribute__((ext_vector_type(4)))  int i32x4;

#ifndef __has_builtin
#define __has_builtin(x) 0
#endif

#if __has_builtin(__builtin_amdgcn_global_load_async_to_lds_b128)
#define HAVE_ASYNC_LDS 1
#else
#define HAVE_ASYNC_LDS 0
#endif

struct FragBits { u32x4 lo, hi; };

// A-matrix 16x32 bf16 fragment: lane l16 holds row M=l16.
// half==0: K = 0..15; half==1: K = 16..31 (contiguous per half-wave).
// p must point to rowBase + khA (khA = half*16); loads p[0..15].
__device__ __forceinline__ v16bf load_frag_a(const bf16* p) {
    FragBits fb;
    fb.lo = *(const u32x4*)(p);
    fb.hi = *(const u32x4*)(p + 8);
    return __builtin_bit_cast(v16bf, fb);
}

// B-matrix 32x16 bf16 fragment: lane l16 holds column N=l16.
// half==0: K = 0..15; half==1: K = 16..31 (contiguous per half-wave).
// p must point to colBase + half*16; loads p[0..15].
__device__ __forceinline__ v16bf load_frag_b(const bf16* p) {
    FragBits fb;
    fb.lo = *(const u32x4*)(p);
    fb.hi = *(const u32x4*)(p + 8);
    return __builtin_bit_cast(v16bf, fb);
}

// 16B global -> LDS stage: async data-mover path on CDNA5, plain copy fallback.
__device__ __forceinline__ void stage16(const bf16* gsrc, bf16* ldst) {
#if HAVE_ASYNC_LDS
    __builtin_amdgcn_global_load_async_to_lds_b128(
        (__attribute__((address_space(1))) i32x4*)gsrc,
        (__attribute__((address_space(3))) i32x4*)ldst, 0, 0);
#else
    *(u32x4*)ldst = *(const u32x4*)gsrc;
#endif
}

__device__ __forceinline__ void wait_async_stage() {
#if HAVE_ASYNC_LDS
#if __has_builtin(__builtin_amdgcn_s_wait_asynccnt)
    __builtin_amdgcn_s_wait_asynccnt(0);
#else
    asm volatile("s_wait_asynccnt 0x0" ::: "memory");
#endif
#endif
}

// DPP16 lane exchange (pure VALU, no LDS): used for 16-lane row reductions.
template <int CTRL>
__device__ __forceinline__ float dpp_xchg(float x) {
    return __builtin_bit_cast(float,
        __builtin_amdgcn_update_dpp(0, __builtin_bit_cast(int, x),
                                    CTRL, 0xF, 0xF, true));
}
// Butterfly reduce across the 16-lane row: quad swaps, then half/full mirrors.
__device__ __forceinline__ float row_max16(float t) {
    t = fmaxf(t, dpp_xchg<0x0B1>(t));   // quad_perm(1,0,3,2)
    t = fmaxf(t, dpp_xchg<0x04E>(t));   // quad_perm(2,3,0,1)
    t = fmaxf(t, dpp_xchg<0x141>(t));   // ROW_HALF_MIRROR
    t = fmaxf(t, dpp_xchg<0x140>(t));   // ROW_MIRROR
    return t;
}
__device__ __forceinline__ float row_sum16(float t) {
    t += dpp_xchg<0x0B1>(t);
    t += dpp_xchg<0x04E>(t);
    t += dpp_xchg<0x141>(t);
    t += dpp_xchg<0x140>(t);
    return t;
}

// ------------------------- format conversion kernels -----------------------

__global__ void cvt_f32_to_bf16(const float* __restrict__ in, bf16* __restrict__ out, int n) {
    int i = blockIdx.x * 256 + threadIdx.x;
    if (i < n) out[i] = (bf16)in[i];
}

// WT[n*D + k] = W[k*D + n]  (weights stored [in,out] -> transposed bf16 [out,in])
__global__ void transpose_cvt_bf16(const float* __restrict__ W, bf16* __restrict__ WT, int D) {
    int idx = blockIdx.x * 256 + threadIdx.x;
    if (idx < D * D) {
        int n = idx / D, k = idx - n * D;
        WT[idx] = (bf16)W[(size_t)k * D + n];
    }
}

// ------------------------------ WMMA GEMM ----------------------------------
// C[M,N] = A[M,K](bf16) @ B[K,N] + bias, with B given transposed: BT[N,K] bf16.
// MODE 0: store bf16 row-major.  MODE 1: store bf16 transposed ([N,M]).
// MODE 2: store f32 row-major with residual add (out = c + bias + resid).
// Block: 256 threads = 8 waves; each wave owns a 16(M) x 64(N) strip.
template <int MODE>
__global__ __launch_bounds__(256, 1)
void gemm_bf16_wmma(const bf16* __restrict__ A, const bf16* __restrict__ BT,
                    const float* __restrict__ bias, const float* __restrict__ resid,
                    void* __restrict__ outp, int M, int Nn, int K) {
    const int lane = threadIdx.x & 31;
    const int wave = threadIdx.x >> 5;
    const int l16  = lane & 15;
    const int hl   = lane >> 4;
    const int khA  = hl * 16;
    const int mBase = blockIdx.x * 64  + (wave & 3) * 16;
    const int nBase = blockIdx.y * 128 + (wave >> 2) * 64;

    v8f acc[4] = {};
    const bf16* arow = A + (size_t)(mBase + l16) * K + khA;

    for (int kb = 0; kb < K; kb += 32) {
        if (kb + 64 < K) __builtin_prefetch(arow + kb + 64, 0, 0);
        v16bf a = load_frag_a(arow + kb);
#pragma unroll
        for (int t = 0; t < 4; ++t) {
            const bf16* bp = BT + (size_t)(nBase + t * 16 + l16) * K + kb + hl * 16;
            v16bf b = load_frag_b(bp);
            acc[t] = __builtin_amdgcn_wmma_f32_16x16x32_bf16(
                false, a, false, b, (short)0, acc[t], false, false);
        }
    }

#pragma unroll
    for (int t = 0; t < 4; ++t) {
#pragma unroll
        for (int r = 0; r < 8; ++r) {
            int row = mBase + r + hl * 8;
            int col = nBase + t * 16 + l16;
            float c = acc[t][r] + bias[col];
            if (MODE == 0) {
                ((bf16*)outp)[(size_t)row * Nn + col] = (bf16)c;
            } else if (MODE == 1) {
                ((bf16*)outp)[(size_t)col * M + row] = (bf16)c;
            } else {
                ((float*)outp)[(size_t)row * Nn + col] = c + resid[(size_t)row * Nn + col];
            }
        }
    }
}

// --------------------------- flash attention -------------------------------
// Block: 128 threads = 4 waves; each wave owns a 16-query tile, all waves share
// one head -> the 32-key K/V tiles are staged once into LDS per block.
// Qb,Kb: [N,D] bf16 row-major. Vt: [D,N] bf16 (transposed). Ob: [N,D] bf16.
__global__ __launch_bounds__(128, 1)
void flash_attn_wmma(const bf16* __restrict__ Qb, const bf16* __restrict__ Kb,
                     const bf16* __restrict__ Vt, const int* __restrict__ ids,
                     bf16* __restrict__ Ob) {
    const int tid  = threadIdx.x;
    const int lane = tid & 31;
    const int wave = tid >> 5;                     // 0..3
    const int l16  = lane & 15;
    const int hl   = lane >> 4;
    const int khA  = hl * 16;
    const int i    = blockIdx.x * 4 + wave;        // query tile (16 rows)
    const int hoff = blockIdx.y * DHEAD;
    const float scale = 0.08838834764831843f;      // 1/sqrt(128)

    __shared__ __align__(16) bf16 ldsK[32 * DHEAD];   // [key 0..31][dh 0..127]
    __shared__ __align__(16) bf16 ldsV[DHEAD * 32];   // [dh 0..127][key 0..31]
    __shared__ __align__(16) bf16 ldsP[4][16 * 32];   // per-wave P scratch
    bf16* lp = &ldsP[wave][0];

    // Q fragments for the whole head dim (4 x K=32)
    v16bf qf[4];
    const bf16* qrow = Qb + (size_t)(i * 16 + l16) * DMODEL + hoff + khA;
#pragma unroll
    for (int kb = 0; kb < 4; ++kb) qf[kb] = load_frag_a(qrow + kb * 32);

    // community ids of this lane's 8 accumulator rows
    int qid[8];
#pragma unroll
    for (int r = 0; r < 8; ++r) qid[r] = ids[i * 16 + r + hl * 8];

    v8f acc[8] = {};
    float mrow[8], lrow[8];
#pragma unroll
    for (int r = 0; r < 8; ++r) { mrow[r] = -__builtin_inff(); lrow[r] = 0.f; }

    for (int jb = 0; jb < NTOK; jb += 32) {
        __syncthreads();   // previous K/V tiles fully consumed
        // ---- stage K (32x128) and V (128x32) tiles: 16B chunks per thread ----
#pragma unroll
        for (int j = 0; j < 4; ++j) {
            int c  = tid + 128 * j;                   // 0..511
            int kr = c >> 4, ko = (c & 15) * 8;       // K tile: row, elem offset
            stage16(Kb + (size_t)(jb + kr) * DMODEL + hoff + ko,
                    ldsK + kr * DHEAD + ko);
            int vr = c >> 2, vo = (c & 3) * 8;        // V tile: dh row, key offset
            stage16(Vt + (size_t)(hoff + vr) * NTOK + jb + vo,
                    ldsV + vr * 32 + vo);
        }
        wait_async_stage();
        __syncthreads();

        // ---- S = Q @ K^T for 32 keys (two 16x16 tiles) ----
        v8f s0 = {}, s1 = {};
#pragma unroll
        for (int kb = 0; kb < 4; ++kb) {
            v16bf b0 = load_frag_b(ldsK + l16        * DHEAD + kb * 32 + hl * 16);
            v16bf b1 = load_frag_b(ldsK + (16 + l16) * DHEAD + kb * 32 + hl * 16);
            s0 = __builtin_amdgcn_wmma_f32_16x16x32_bf16(false, qf[kb], false, b0, (short)0, s0, false, false);
            s1 = __builtin_amdgcn_wmma_f32_16x16x32_bf16(false, qf[kb], false, b1, (short)0, s1, false, false);
        }

        const int kid0 = ids[jb + l16];
        const int kid1 = ids[jb + 16 + l16];

        // ---- mask + online softmax; rows live across 16-lane DPP rows ----
#pragma unroll
        for (int r = 0; r < 8; ++r) {
            float v0 = s0[r] * scale; if (qid[r] == kid0) v0 = -__builtin_inff();
            float v1 = s1[r] * scale; if (qid[r] == kid1) v1 = -__builtin_inff();
            float t    = row_max16(fmaxf(v0, v1));
            float mold = mrow[r];
            float mnew = fmaxf(mold, t);
            bool  dead = (mnew == -__builtin_inff());
            float corr = dead ? 1.f : __expf(mold - mnew);
            float p0   = dead ? 0.f : __expf(v0 - mnew);
            float p1   = dead ? 0.f : __expf(v1 - mnew);
            lrow[r] = lrow[r] * corr + row_sum16(p0 + p1);
            mrow[r] = mnew;
#pragma unroll
            for (int t2 = 0; t2 < 8; ++t2) acc[t2][r] *= corr;
            int rr = r + hl * 8;                    // C-layout row -> LDS row
            lp[rr * 32 + l16]      = (bf16)p0;
            lp[rr * 32 + 16 + l16] = (bf16)p1;
        }
        v16bf pf = load_frag_a(lp + l16 * 32 + khA);  // same-wave LDS: in-order

        // ---- O += P @ V  (8 column tiles of 16) ----
#pragma unroll
        for (int t = 0; t < 8; ++t) {
            v16bf vb = load_frag_b(ldsV + (t * 16 + l16) * 32 + hl * 16);
            acc[t] = __builtin_amdgcn_wmma_f32_16x16x32_bf16(false, pf, false, vb, (short)0, acc[t], false, false);
        }
    }

    // ---- normalize and store ----
#pragma unroll
    for (int t = 0; t < 8; ++t) {
#pragma unroll
        for (int r = 0; r < 8; ++r) {
            int row = i * 16 + r + hl * 8;
            int col = hoff + t * 16 + l16;
            float lv = lrow[r];
            float o = (lv > 0.f) ? (acc[t][r] / lv) : 0.f;
            Ob[(size_t)row * DMODEL + col] = (bf16)o;
        }
    }
}

// ----------------------------- layernorm -----------------------------------
// In-place over d_out rows (one block per row; block owns the row exclusively).
__global__ void layernorm_inplace(float* __restrict__ out, const float* __restrict__ gamma,
                                  const float* __restrict__ beta) {
    const int row = blockIdx.x;
    const int tid = threadIdx.x;
    __shared__ float red[256];
    float* rp = out + (size_t)row * DMODEL;

    float v[4];
    float s = 0.f;
#pragma unroll
    for (int j = 0; j < 4; ++j) { v[j] = rp[tid + j * 256]; s += v[j]; }
    red[tid] = s; __syncthreads();
    for (int k = 128; k > 0; k >>= 1) { if (tid < k) red[tid] += red[tid + k]; __syncthreads(); }
    const float mu = red[0] * (1.0f / DMODEL);
    __syncthreads();

    float vs = 0.f;
#pragma unroll
    for (int j = 0; j < 4; ++j) { float d = v[j] - mu; vs += d * d; }
    red[tid] = vs; __syncthreads();
    for (int k = 128; k > 0; k >>= 1) { if (tid < k) red[tid] += red[tid + k]; __syncthreads(); }
    const float rstd = rsqrtf(red[0] * (1.0f / DMODEL) + 1e-5f);

#pragma unroll
    for (int j = 0; j < 4; ++j) {
        int c = tid + j * 256;
        rp[c] = (v[j] - mu) * rstd * gamma[c] + beta[c];
    }
}

// ------------------------------- launcher ----------------------------------

extern "C" void kernel_launch(void* const* d_in, const int* in_sizes, int n_in,
                              void* d_out, int out_size, void* d_ws, size_t ws_size,
                              hipStream_t stream) {
    const float* x     = (const float*)d_in[0];
    const int*   ids   = (const int*)  d_in[1];
    const float* Wq    = (const float*)d_in[2];
    const float* bq    = (const float*)d_in[3];
    const float* Wk    = (const float*)d_in[4];
    const float* bk    = (const float*)d_in[5];
    const float* Wv    = (const float*)d_in[6];
    const float* bv    = (const float*)d_in[7];
    const float* Wo    = (const float*)d_in[8];
    const float* bo    = (const float*)d_in[9];
    const float* gamma = (const float*)d_in[10];
    const float* beta  = (const float*)d_in[11];
    float* out = (float*)d_out;

    const size_t ND = (size_t)NTOK * DMODEL;
    const size_t DD = (size_t)DMODEL * DMODEL;

    bf16* xb  = (bf16*)d_ws;
    bf16* WqT = xb  + ND;
    bf16* WkT = WqT + DD;
    bf16* WvT = WkT + DD;
    bf16* WoT = WvT + DD;
    bf16* Qb  = WoT + DD;
    bf16* Kb  = Qb  + ND;
    bf16* Vt  = Kb  + ND;   // stored [D, N]
    bf16* Ob  = Vt  + ND;

    cvt_f32_to_bf16<<<(int)(ND / 256), 256, 0, stream>>>(x, xb, (int)ND);
    transpose_cvt_bf16<<<(int)(DD / 256), 256, 0, stream>>>(Wq, WqT, DMODEL);
    transpose_cvt_bf16<<<(int)(DD / 256), 256, 0, stream>>>(Wk, WkT, DMODEL);
    transpose_cvt_bf16<<<(int)(DD / 256), 256, 0, stream>>>(Wv, WvT, DMODEL);
    transpose_cvt_bf16<<<(int)(DD / 256), 256, 0, stream>>>(Wo, WoT, DMODEL);

    dim3 gg(NTOK / 64, DMODEL / 128);
    gemm_bf16_wmma<0><<<gg, 256, 0, stream>>>(xb, WqT, bq, nullptr, Qb, NTOK, DMODEL, DMODEL);
    gemm_bf16_wmma<0><<<gg, 256, 0, stream>>>(xb, WkT, bk, nullptr, Kb, NTOK, DMODEL, DMODEL);
    gemm_bf16_wmma<1><<<gg, 256, 0, stream>>>(xb, WvT, bv, nullptr, Vt, NTOK, DMODEL, DMODEL);

    flash_attn_wmma<<<dim3(NTOK / 16 / 4, NHEAD), 128, 0, stream>>>(Qb, Kb, Vt, ids, Ob);

    gemm_bf16_wmma<2><<<gg, 256, 0, stream>>>(Ob, WoT, bo, x, out, NTOK, DMODEL, DMODEL);
    layernorm_inplace<<<NTOK, 256, 0, stream>>>(out, gamma, beta);
}
